// GATGNN_76184129896628
// MI455X (gfx1250) — matrix operand order repaired and checked by reading, not compiled, not run on hardware
//
#include <hip/hip_runtime.h>
#include <math.h>

typedef __attribute__((ext_vector_type(16))) _Float16 v16h;
typedef __attribute__((ext_vector_type(4)))  _Float16 v4h;
typedef __attribute__((ext_vector_type(8)))  float    v8f;
typedef __attribute__((ext_vector_type(4)))  float    v4f;

#define NEG_SLOPE 0.2f

__device__ inline v4h cvt4(v4f v) {
  v4h r;
  r[0] = (_Float16)v[0]; r[1] = (_Float16)v[1];
  r[2] = (_Float16)v[2]; r[3] = (_Float16)v[3];
  return r;
}

// ---------------- zero fill ----------------
__global__ void k_zero(float* p, int count) {
  int i = blockIdx.x * blockDim.x + threadIdx.x;
  if (i < count) p[i] = 0.0f;
}

// ---------------- WMMA GEMM: C[M,N] = A[M,K] @ B[N,K]^T (+bias) -------------
// 256 threads = 8 wave32; block tile 64(M) x 64(N); K-step 32.
// waves 0-3: M-subtile r=w, N-subtiles {0,1}; waves 4-7: N-subtiles {2,3}.
// A rows are CLAMPED to M-1 (garbage rows are never stored), so the staging
// path is branch-free: 4x global_load_b128 -> cvt -> 4x ds_store_b64.
__global__ __launch_bounds__(256)
void k_gemm_wmma(const float* __restrict__ A, const float* __restrict__ B,
                 const float* __restrict__ bias, float* __restrict__ C,
                 int M, int N, int K) {
  __shared__ _Float16 As[64][36];
  __shared__ _Float16 Bs[64][36];
  const int tid  = threadIdx.x;
  const int lane = tid & 31;
  const int wave = tid >> 5;
  const int hb   = lane >> 4;   // half-wave (selects K-group per ISA layout)
  const int l16  = lane & 15;
  const int blockM = blockIdx.y * 64;
  const int blockN = blockIdx.x * 64;
  const int r  = wave & 3;          // M subtile index
  const int c0 = (wave >> 2) * 2;   // first of two N subtile indices

  // staging coords: 256 threads cover 32 rows x (8 threads * 4 floats)
  const int srow = tid >> 3;         // 0..31
  const int scol = (tid & 7) * 4;    // 0,4,...,28
  const int ar0 = min(blockM + srow,      M - 1);
  const int ar1 = min(blockM + srow + 32, M - 1);
  const int br0 = blockN + srow;          // N dims are multiples of 64
  const int br1 = blockN + srow + 32;

  v8f acc0 = {};
  v8f acc1 = {};

  for (int kb = 0; kb < K; kb += 32) {
    v4f a0 = *(const v4f*)&A[(size_t)ar0 * K + kb + scol];
    v4f a1 = *(const v4f*)&A[(size_t)ar1 * K + kb + scol];
    v4f b0 = *(const v4f*)&B[(size_t)br0 * K + kb + scol];
    v4f b1 = *(const v4f*)&B[(size_t)br1 * K + kb + scol];
    *(v4h*)&As[srow     ][scol] = cvt4(a0);
    *(v4h*)&As[srow + 32][scol] = cvt4(a1);
    *(v4h*)&Bs[srow     ][scol] = cvt4(b0);
    *(v4h*)&Bs[srow + 32][scol] = cvt4(b1);
    __syncthreads();

    // prefetch next K-tile into cache while WMMA phase runs
    if (kb + 32 < K) {
      __builtin_prefetch(&A[(size_t)ar0 * K + kb + 32 + scol], 0, 3);
      __builtin_prefetch(&B[(size_t)br0 * K + kb + 32 + scol], 0, 3);
    }

    // Build fragments per ISA 7.12.2:
    // A 16x32 f16: lane<16 -> K {0..7,16..23}; lane>=16 -> K {8..15,24..31}
    // B 32x16 f16: same K order as A: lane<16 -> K {0..7,16..23}; lane>=16 -> K {8..15,24..31}
    v16h af, bf0, bf1;
    #pragma unroll
    for (int e = 0; e < 16; ++e) {
      int ka  = (e & 7) + ((e >> 3) << 4) + (hb << 3);
      int kbq = (e & 7) + ((e >> 3) << 4) + (hb << 3);
      af[e]  = As[r  * 16 + l16][ka];
      bf0[e] = Bs[c0 * 16 + l16][kbq];
      bf1[e] = Bs[(c0 + 1) * 16 + l16][kbq];
    }
    acc0 = __builtin_amdgcn_wmma_f32_16x16x32_f16(false, af, false, bf0,
                                                  (short)0, acc0, false, false);
    acc1 = __builtin_amdgcn_wmma_f32_16x16x32_f16(false, af, false, bf1,
                                                  (short)0, acc1, false, false);
    __syncthreads();
  }

  // D layout: element i -> row M = i + 8*hb, col N = lane%16
  int coln0 = blockN + c0 * 16 + l16;
  int coln1 = coln0 + 16;
  float bv0 = bias ? bias[coln0] : 0.0f;
  float bv1 = bias ? bias[coln1] : 0.0f;
  #pragma unroll
  for (int i = 0; i < 8; ++i) {
    int row = blockM + r * 16 + i + 8 * hb;
    if (row < M) {
      C[(size_t)row * N + coln0] = acc0[i] + bv0;
      C[(size_t)row * N + coln1] = acc1[i] + bv1;
    }
  }
}

// ------------- attention coefficients: a = einsum('nhc,hc->nh') -------------
template <int HEADS, int HID>
__global__ void k_att_scores(const float* __restrict__ h,
                             const float* __restrict__ att_src,
                             const float* __restrict__ att_dst,
                             float* __restrict__ a_src, float* __restrict__ a_dst,
                             int n) {
  int gw   = (blockIdx.x * blockDim.x + threadIdx.x) >> 5;
  int lane = threadIdx.x & 31;
  if (gw >= n * HEADS) return;
  int node = gw / HEADS, hd = gw % HEADS;
  const float* hp = h + (size_t)node * HEADS * HID + hd * HID;
  float ss = 0.f, sd = 0.f;
  for (int c = lane; c < HID; c += 32) {
    float v = hp[c];
    ss += v * att_src[hd * HID + c];
    sd += v * att_dst[hd * HID + c];
  }
  for (int off = 16; off; off >>= 1) {
    ss += __shfl_xor(ss, off, 32);
    sd += __shfl_xor(sd, off, 32);
  }
  if (lane == 0) { a_src[gw] = ss; a_dst[gw] = sd; }
}

// order-preserving float<->uint for atomic segment-max
__device__ inline unsigned f2sort(float f) {
  unsigned u = __float_as_uint(f);
  return (u & 0x80000000u) ? ~u : (u | 0x80000000u);
}
__device__ inline float sort2f(unsigned u) {
  unsigned v = (u & 0x80000000u) ? (u & 0x7fffffffu) : ~u;
  return __uint_as_float(v);
}

template <int HEADS>
__global__ void k_edge_alpha_max(const int* __restrict__ ei, int E, int Etot,
                                 const float* __restrict__ a_src,
                                 const float* __restrict__ a_dst,
                                 float* __restrict__ alpha,
                                 unsigned* __restrict__ amax) {
  int idx = blockIdx.x * blockDim.x + threadIdx.x;
  if (idx >= Etot * HEADS) return;
  int e = idx / HEADS, hd = idx % HEADS;
  int s, d;
  if (e < E) { s = ei[e]; d = ei[E + e]; } else { s = d = e - E; }
  float a = a_src[s * HEADS + hd] + a_dst[d * HEADS + hd];
  a = (a > 0.f) ? a : NEG_SLOPE * a;          // leaky_relu
  alpha[idx] = a;
  atomicMax(&amax[d * HEADS + hd], f2sort(a));
}

template <int HEADS>
__global__ void k_edge_exp_sum(const int* __restrict__ ei, int E, int Etot,
                               float* __restrict__ alpha,
                               const unsigned* __restrict__ amax,
                               float* __restrict__ denom) {
  int idx = blockIdx.x * blockDim.x + threadIdx.x;
  if (idx >= Etot * HEADS) return;
  int e = idx / HEADS, hd = idx % HEADS;
  int d = (e < E) ? ei[E + e] : (e - E);
  float mu = sort2f(amax[d * HEADS + hd]);
  if (!__builtin_isfinite(mu)) mu = 0.f;      // matches where(isfinite, amax, 0)
  float ex = __expf(alpha[idx] - mu);
  alpha[idx] = ex;                             // reuse buffer for ex
  atomicAdd(&denom[d * HEADS + hd], ex);
}

// wave per edge; lanes stride channels; L2-resident f32 scatter-adds
template <int HEADS, int HID>
__global__ void k_edge_aggregate(const int* __restrict__ ei, int E, int Etot,
                                 const float* __restrict__ ex,
                                 const float* __restrict__ denom,
                                 const float* __restrict__ h,
                                 float* __restrict__ agg) {
  int e    = (blockIdx.x * blockDim.x + threadIdx.x) >> 5;
  int lane = threadIdx.x & 31;
  if (e >= Etot) return;
  int s, d;
  if (e < E) { s = ei[e]; d = ei[E + e]; } else { s = d = e - E; }
  const float* hs = h + (size_t)s * HEADS * HID;
  float* ad = agg + (size_t)d * HEADS * HID;
  #pragma unroll
  for (int hd = 0; hd < HEADS; ++hd) {
    float w = ex[e * HEADS + hd] / (denom[d * HEADS + hd] + 1e-16f);
    for (int c = lane; c < HID; c += 32)
      atomicAdd(&ad[hd * HID + c], w * hs[hd * HID + c]);
  }
}

// dim is a power of two -> mask instead of %
__global__ void k_bias_elu(float* x, const float* __restrict__ bias,
                           int count, int dimMask) {
  int i = blockIdx.x * blockDim.x + threadIdx.x;
  if (i >= count) return;
  float v = x[i] + bias[i & dimMask];
  x[i] = (v > 0.f) ? v : (__expf(v) - 1.f);
}

__global__ void k_edge_scores(const float* __restrict__ h,
                              const float* __restrict__ Wedge,
                              float* __restrict__ s1, float* __restrict__ s2,
                              int n, int hid) {
  int gw   = (blockIdx.x * blockDim.x + threadIdx.x) >> 5;
  int lane = threadIdx.x & 31;
  if (gw >= n) return;
  const float* hp = h + (size_t)gw * hid;
  float v1 = 0.f, v2 = 0.f;
  for (int c = lane; c < hid; c += 32) {
    float v = hp[c];
    v1 += v * Wedge[c];
    v2 += v * Wedge[hid + c];
  }
  for (int off = 16; off; off >>= 1) {
    v1 += __shfl_xor(v1, off, 32);
    v2 += __shfl_xor(v2, off, 32);
  }
  if (lane == 0) { s1[gw] = v1; s2[gw] = v2; }
}

// 400 MB streaming writer: v4f non-temporal stores (NT hint keeps L2 clean)
__global__ void k_outer(const float* __restrict__ s1, const float* __restrict__ s2,
                        const float* __restrict__ bedge, float* __restrict__ out,
                        int n) {
  int j4 = blockIdx.x * blockDim.x + threadIdx.x;
  int i  = blockIdx.y;
  int nq = n >> 2;
  if (j4 >= nq) return;
  float base = s1[i] + bedge[0];
  v4f   sv   = ((const v4f*)s2)[j4];
  v4f   o    = sv + base;
  __builtin_nontemporal_store(o, ((v4f*)(out + (size_t)i * n)) + j4);
}

extern "C" void kernel_launch(void* const* d_in, const int* in_sizes, int n_in,
                              void* d_out, int out_size, void* d_ws, size_t ws_size,
                              hipStream_t stream) {
  const float* x    = (const float*)d_in[0];
  const int*   ei   = (const int*)d_in[1];
  const float* W0   = (const float*)d_in[2];
  const float* as0  = (const float*)d_in[3];
  const float* ad0  = (const float*)d_in[4];
  const float* b0   = (const float*)d_in[5];
  const float* W1   = (const float*)d_in[6];
  const float* as1  = (const float*)d_in[7];
  const float* ad1  = (const float*)d_in[8];
  const float* b1   = (const float*)d_in[9];
  const float* Wout = (const float*)d_in[10];
  const float* bout = (const float*)d_in[11];
  const float* We   = (const float*)d_in[12];
  const float* be   = (const float*)d_in[13];

  const int IN = 128, HID = 128, HEADS = 4;
  const int n    = in_sizes[0] / IN;
  const int E    = in_sizes[1] / 2;
  const int Etot = E + n;

  float* ws = (float*)d_ws;
  size_t off = 0;
  auto alloc = [&](size_t cnt) { float* p = ws + off; off += cnt; return p; };
  float*    h0    = alloc((size_t)n * HEADS * HID);
  float*    agg0  = alloc((size_t)n * HEADS * HID);
  float*    h1    = alloc((size_t)n * HID);
  float*    agg1  = alloc((size_t)n * HID);
  float*    hout  = alloc((size_t)n * HID);
  float*    asrc0 = alloc((size_t)n * HEADS);
  float*    adst0 = alloc((size_t)n * HEADS);
  unsigned* amax0 = (unsigned*)alloc((size_t)n * HEADS);
  float*    den0  = alloc((size_t)n * HEADS);      // contiguous after amax0
  float*    asrc1 = alloc((size_t)n);
  float*    adst1 = alloc((size_t)n);
  unsigned* amax1 = (unsigned*)alloc((size_t)n);
  float*    den1  = alloc((size_t)n);              // contiguous after amax1
  float*    s1    = alloc((size_t)n);
  float*    s2    = alloc((size_t)n);
  float*    alp0  = alloc((size_t)Etot * HEADS);
  float*    alp1  = alloc((size_t)Etot);

  // ---------- layer 0 (heads=4) ----------
  k_gemm_wmma<<<dim3((HEADS * HID) / 64, (n + 63) / 64), 256, 0, stream>>>(
      x, W0, nullptr, h0, n, HEADS * HID, IN);
  k_att_scores<4, 128><<<(n * HEADS * 32 + 255) / 256, 256, 0, stream>>>(
      h0, as0, ad0, asrc0, adst0, n);
  k_zero<<<(n * HEADS * 2 + 255) / 256, 256, 0, stream>>>((float*)amax0, n * HEADS * 2);
  k_zero<<<(n * HEADS * HID + 255) / 256, 256, 0, stream>>>(agg0, n * HEADS * HID);
  k_edge_alpha_max<4><<<(Etot * HEADS + 255) / 256, 256, 0, stream>>>(
      ei, E, Etot, asrc0, adst0, alp0, amax0);
  k_edge_exp_sum<4><<<(Etot * HEADS + 255) / 256, 256, 0, stream>>>(
      ei, E, Etot, alp0, amax0, den0);
  k_edge_aggregate<4, 128><<<(Etot * 32 + 255) / 256, 256, 0, stream>>>(
      ei, E, Etot, alp0, den0, h0, agg0);
  k_bias_elu<<<(n * HEADS * HID + 255) / 256, 256, 0, stream>>>(
      agg0, b0, n * HEADS * HID, HEADS * HID - 1);

  // ---------- layer 1 (heads=1) ----------
  k_gemm_wmma<<<dim3(HID / 64, (n + 63) / 64), 256, 0, stream>>>(
      agg0, W1, nullptr, h1, n, HID, HEADS * HID);
  k_att_scores<1, 128><<<(n * 32 + 255) / 256, 256, 0, stream>>>(
      h1, as1, ad1, asrc1, adst1, n);
  k_zero<<<(n * 2 + 255) / 256, 256, 0, stream>>>((float*)amax1, n * 2);
  k_zero<<<(n * HID + 255) / 256, 256, 0, stream>>>(agg1, n * HID);
  k_edge_alpha_max<1><<<(Etot + 255) / 256, 256, 0, stream>>>(
      ei, E, Etot, asrc1, adst1, alp1, amax1);
  k_edge_exp_sum<1><<<(Etot + 255) / 256, 256, 0, stream>>>(
      ei, E, Etot, alp1, amax1, den1);
  k_edge_aggregate<1, 128><<<(Etot * 32 + 255) / 256, 256, 0, stream>>>(
      ei, E, Etot, alp1, den1, h1, agg1);
  k_bias_elu<<<(n * HID + 255) / 256, 256, 0, stream>>>(
      agg1, b1, n * HID, HID - 1);

  // ---------- output projection (bias fused) ----------
  k_gemm_wmma<<<dim3(HID / 64, (n + 63) / 64), 256, 0, stream>>>(
      agg1, Wout, bout, hout, n, HID, HID);

  // ---------- edge logits ----------
  k_edge_scores<<<(n * 32 + 255) / 256, 256, 0, stream>>>(hout, We, s1, s2, n, HID);
  k_outer<<<dim3(((n / 4) + 255) / 256, n), 256, 0, stream>>>(
      s1, s2, be, (float*)d_out, n);
}
